// GCN2_25056839205778
// MI455X (gfx1250) — matrix operand order repaired and checked by reading, not compiled, not run on hardware
//
#include <hip/hip_runtime.h>
#include <hip/hip_bf16.h>

// ---------------------------------------------------------------------------
// GCN 2-layer inference for MI455X (gfx1250, wave32, WMMA).
//   L1: support1 = bf16(x) @ bf16(W1)      -> v_wmma_f32_16x16x32_bf16
//       agg1     = b1 + scatter_add(support1[src]*ew -> dst)   (f32 atomics)
//       h        = bf16(relu(agg1))
//   L2: support2 = h @ bf16(W2 padded 40->48)
//       out      = b2 + scatter_add(support2[src]*ew -> dst)
// ---------------------------------------------------------------------------

typedef __attribute__((ext_vector_type(16))) __bf16        v16bf;
typedef __attribute__((ext_vector_type(8)))  float         v8f;
typedef __attribute__((ext_vector_type(4)))  unsigned int  u32x4;

#define N_NODES   100000
#define N_EDGES   1600000
#define NFEAT     256
#define NHID      128
#define NCLASS    40
#define NCLASS_P  48   // padded to 3 N-tiles of 16

union FragBf16 {            // 16 bf16 = 32 bytes = two b128 loads
    u32x4 u[2];
    v16bf v;
};

__device__ __forceinline__ unsigned short f2bf_rne(float f) {
    union { float f; unsigned int u; } x; x.f = f;
    unsigned int u = x.u;
    if ((u & 0x7fffffffu) > 0x7f800000u)            // NaN -> quiet NaN bf16
        return (unsigned short)((u >> 16) | 0x0040u);
    return (unsigned short)((u + 0x7fffu + ((u >> 16) & 1u)) >> 16);
}

// ---- elementwise f32 -> bf16 ------------------------------------------------
__global__ void gcn_cvt_bf16(const float* __restrict__ in,
                             unsigned short* __restrict__ out, int n) {
    int i = blockIdx.x * blockDim.x + threadIdx.x;
    if (i < n) out[i] = f2bf_rne(in[i]);
}

// ---- fused bias-add(agg init) + relu + bf16 convert -------------------------
__global__ void gcn_relu_cvt_bf16(const float* __restrict__ in,
                                  unsigned short* __restrict__ out, int n) {
    int i = blockIdx.x * blockDim.x + threadIdx.x;
    if (i < n) out[i] = f2bf_rne(fmaxf(in[i], 0.0f));
}

// ---- broadcast bias into accumulation buffer --------------------------------
__global__ void gcn_init_bias(const float* __restrict__ b,
                              float* __restrict__ out, int F, int total) {
    int i = blockIdx.x * blockDim.x + threadIdx.x;
    if (i < total) out[i] = b[i % F];
}

// ---- pack W (KxN f32, row-major) into WMMA B-fragment order -----------------
// Bpack[((kt*NT + nt)*32 + lane)*16 + j] = W[kt*32 + (lane>>4)*16 + j][nt*16 + (lane&15)]
__global__ void gcn_pack_w(const float* __restrict__ W,
                           unsigned short* __restrict__ Wp,
                           int K, int N, int NT, int total) {
    int idx = blockIdx.x * blockDim.x + threadIdx.x;
    if (idx >= total) return;
    int j    = idx & 15;
    int rest = idx >> 4;
    int l    = rest & 31;
    rest   >>= 5;
    int nt   = rest % NT;
    int kt   = rest / NT;
    int k    = kt * 32 + (l >> 4) * 16 + j;
    int n    = nt * 16 + (l & 15);
    Wp[idx]  = (k < K && n < N) ? f2bf_rne(W[(size_t)k * N + n]) : (unsigned short)0;
}

// ---- WMMA GEMM: C[M x NT*16] = A[M x K](bf16) * Bpack ----------------------
// One block per 16-row M tile (M = 6250*16 exactly). NT waves per block; wave w
// owns N-tile w. The 16xK A tile is contiguous in memory -> coalesced LDS fill.
template <int K, int NT>
__global__ __launch_bounds__(32 * NT)
void gcn_gemm_wmma_bf16(const unsigned short* __restrict__ A,
                        const unsigned short* __restrict__ Bpack,
                        float* __restrict__ C) {
    __shared__ unsigned short Atile[16 * K];

    const int tid  = threadIdx.x;
    const int lane = tid & 31;
    const int wave = tid >> 5;

    // Stage contiguous 16xK bf16 tile into LDS (b128 chunks of 8 halves).
    const u32x4* gsrc = (const u32x4*)(A + (size_t)blockIdx.x * 16 * K);
    u32x4*       ldst = (u32x4*)Atile;
    constexpr int NVEC = (16 * K) / 8;
    for (int i = tid; i < NVEC; i += 32 * NT) ldst[i] = gsrc[i];
    __syncthreads();

    const int row = lane & 15;   // M within tile
    const int hk  = lane >> 4;   // K-half select

    v8f acc = {};
    constexpr int KT = K / 32;
    for (int kt = 0; kt < KT; ++kt) {
        // A fragment, contiguous K order matching Bpack:
        //   VGPR0-3: K = hk*16 + 0..7 ; VGPR4-7: K = hk*16 + 8..15
        FragBf16 a;
        a.u[0] = *(const u32x4*)(Atile + row * K + kt * 32 + hk * 16);
        a.u[1] = *(const u32x4*)(Atile + row * K + kt * 32 + hk * 16 + 8);

        // B fragment: pre-packed, 32B contiguous per lane (1KB/wave, coalesced)
        FragBf16 b;
        const u32x4* bp = (const u32x4*)(Bpack + ((size_t)(kt * NT + wave) * 32 + lane) * 16);
        b.u[0] = bp[0];
        b.u[1] = bp[1];

        acc = __builtin_amdgcn_wmma_f32_16x16x32_bf16(
                  false, a.v, false, b.v, (short)0, acc, false, false);
    }

    // C/D layout: VGPR r -> M = r (lanes 0-15) / M = 8+r (lanes 16-31), N = lane&15
    const int colBase = wave * 16 + (lane & 15);
    float* crow = C + ((size_t)blockIdx.x * 16 + hk * 8) * (NT * 16) + colBase;
#pragma unroll
    for (int r = 0; r < 8; ++r)
        crow[(size_t)r * (NT * 16)] = acc[r];
}

// ---- edge gather/scale/scatter-add (L2-resident f32 atomics) ----------------
// F = output feature count used; SUP = row stride of support buffer.
template <int F, int SUP>
__global__ void gcn_scatter_edges(const float* __restrict__ sup,
                                  const long long* __restrict__ srcIdx,
                                  const long long* __restrict__ dstIdx,
                                  const float* __restrict__ ew,
                                  float* __restrict__ agg) {
    int t = blockIdx.x * blockDim.x + threadIdx.x;
    int e = t / F;
    int f = t - e * F;
    if (e >= N_EDGES) return;
    int s = (int)srcIdx[e];
    int d = (int)dstIdx[e];
    float v = sup[(size_t)s * SUP + f] * ew[e];
    unsafeAtomicAdd(&agg[(size_t)d * F + f], v);   // global_atomic_add_f32
}

// ---------------------------------------------------------------------------
extern "C" void kernel_launch(void* const* d_in, const int* in_sizes, int n_in,
                              void* d_out, int out_size, void* d_ws, size_t ws_size,
                              hipStream_t stream) {
    const float*     x    = (const float*)d_in[0];
    const long long* eidx = (const long long*)d_in[1];   // int64 [2, E]
    const float*     ew   = (const float*)d_in[2];
    const float*     W1   = (const float*)d_in[3];
    const float*     b1   = (const float*)d_in[4];
    const float*     W2   = (const float*)d_in[5];
    const float*     b2   = (const float*)d_in[6];
    float*           out  = (float*)d_out;

    const long long* srcI = eidx;
    const long long* dstI = eidx + N_EDGES;

    // workspace carve-out (all sizes 256B-multiples)
    char*  ws  = (char*)d_ws;
    size_t off = 0;
    auto carve = [&](size_t bytes) {
        void* p = ws + off;
        off += (bytes + 255) & ~(size_t)255;
        return p;
    };
    unsigned short* xb   = (unsigned short*)carve((size_t)N_NODES * NFEAT * 2);    // 51.2 MB
    float*          sup1 = (float*)         carve((size_t)N_NODES * NHID  * 4);    // 51.2 MB
    float*          agg1 = (float*)         carve((size_t)N_NODES * NHID  * 4);    // 51.2 MB
    unsigned short* hb   = (unsigned short*)carve((size_t)N_NODES * NHID  * 2);    // 25.6 MB
    float*          sup2 = (float*)         carve((size_t)N_NODES * NCLASS_P * 4); // 19.2 MB
    unsigned short* W1p  = (unsigned short*)carve((size_t)8 * 8 * 512 * 2);        // 64 KB
    unsigned short* W2p  = (unsigned short*)carve((size_t)4 * 3 * 512 * 2);        // 12 KB

    const int nX  = N_NODES * NFEAT;   // 25,600,000
    const int nH  = N_NODES * NHID;    // 12,800,000
    const int nO  = N_NODES * NCLASS;  //  4,000,000
    const int pw1 = 8 * 8 * 512;       // 32,768 packed W1 elements
    const int pw2 = 4 * 3 * 512;       //  6,144 packed W2 elements

    // --- precision conversion / weight packing ---
    gcn_cvt_bf16<<<(nX + 255) / 256, 256, 0, stream>>>(x, xb, nX);
    gcn_pack_w<<<(pw1 + 255) / 256, 256, 0, stream>>>(W1, W1p, NFEAT, NHID,   8, pw1);
    gcn_pack_w<<<(pw2 + 255) / 256, 256, 0, stream>>>(W2, W2p, NHID,  NCLASS, 3, pw2);

    // --- layer 1 ---
    gcn_gemm_wmma_bf16<NFEAT, 8><<<N_NODES / 16, 256, 0, stream>>>(xb, W1p, sup1);
    gcn_init_bias<<<(nH + 255) / 256, 256, 0, stream>>>(b1, agg1, NHID, nH);
    {
        long long tot = (long long)N_EDGES * NHID;           // 204.8 M
        gcn_scatter_edges<NHID, NHID>
            <<<(unsigned)((tot + 255) / 256), 256, 0, stream>>>(sup1, srcI, dstI, ew, agg1);
    }
    gcn_relu_cvt_bf16<<<(nH + 255) / 256, 256, 0, stream>>>(agg1, hb, nH);

    // --- layer 2 ---
    gcn_gemm_wmma_bf16<NHID, 3><<<N_NODES / 16, 96, 0, stream>>>(hb, W2p, sup2);
    gcn_init_bias<<<(nO + 255) / 256, 256, 0, stream>>>(b2, out, NCLASS, nO);
    {
        long long tot = (long long)N_EDGES * NCLASS;         // 64 M
        gcn_scatter_edges<NCLASS, NCLASS_P>
            <<<(unsigned)((tot + 255) / 256), 256, 0, stream>>>(sup2, srcI, dstI, ew, out);
    }
}
